// CrossAttention_31001073943289
// MI455X (gfx1250) — matrix operand order repaired and checked by reading, not compiled, not run on hardware
//
#include <hip/hip_runtime.h>
#include <math.h>

// ---------------------------------------------------------------------------
// Cross-attention for MI455X (gfx1250), bf16 WMMA + flash-attention streaming.
// B=4, C=256, CK=32, N=4096.  All matmuls via v_wmma_f32_16x16x32_bf16.
// Workspace layout (bytes):
//   [0,      1MB)  Qb  : bf16 [B][N][CK]   (q projection, row-major over CK)
//   [1MB,    2MB)  Kb  : bf16 [B][N][CK]   (k^T projection)
//   [2MB,   10MB)  Vt  : bf16 [B][C][N]    (v projection, channel-major)
//   [10MB, +16KB)  Wq bf16, then Wk bf16 (16KB), then Wv bf16 (128KB)
// Total ~10.16MB; everything fits in the 192MB L2 so the attention pass
// streams K/V from L2 rather than HBM.
// ---------------------------------------------------------------------------

typedef __attribute__((ext_vector_type(16))) __bf16 v16bf;
typedef __attribute__((ext_vector_type(8)))  __bf16 v8bf;
typedef __attribute__((ext_vector_type(8)))  float  v8f;

#define B_  4
#define C_  256
#define CK_ 32
#define N_  4096

static __device__ __forceinline__ v8f wmma_bf16(v16bf a, v16bf b, v8f c) {
  return __builtin_amdgcn_wmma_f32_16x16x32_bf16(false, a, false, b,
                                                 (short)0, c, false, false);
}

static __device__ __forceinline__ v16bf cat8(v8bf lo, v8bf hi) {
  v16bf r;
#pragma unroll
  for (int i = 0; i < 8; ++i) { r[i] = lo[i]; r[i + 8] = hi[i]; }
  return r;
}

// A-matrix 16x32 bf16 fragment. rowptr points at k=0 of this lane's row (m=lane%16).
// element e <-> k = 16*hh + e  (hh = lane/16)
static __device__ __forceinline__ v16bf load_a16(const __bf16* rowptr, int hh) {
  v8bf lo = *(const v8bf*)(rowptr + hh * 16);
  v8bf hi = *(const v8bf*)(rowptr + hh * 16 + 8);
  return cat8(lo, hi);
}

// B-matrix 32x16 bf16 fragment; colptr points at this lane's 16 consecutive
// k values (caller already added hh*16).  element e <-> k = 16*hh + e.
static __device__ __forceinline__ v16bf load_b16(const __bf16* colptr) {
  v8bf lo = *(const v8bf*)(colptr);
  v8bf hi = *(const v8bf*)(colptr + 8);
  return cat8(lo, hi);
}

// ---------------------------------------------------------------------------
// Kernel 1: convert weights f32 -> bf16 into workspace.
// ---------------------------------------------------------------------------
__global__ void convert_w_kernel(const float* __restrict__ Wq,
                                 const float* __restrict__ Wk,
                                 const float* __restrict__ Wv,
                                 __bf16* __restrict__ wq, __bf16* __restrict__ wk,
                                 __bf16* __restrict__ wv) {
  int i = blockIdx.x * blockDim.x + threadIdx.x;
  if (i < CK_ * C_) { wq[i] = (__bf16)Wq[i]; wk[i] = (__bf16)Wk[i]; }
  if (i < C_ * C_)  { wv[i] = (__bf16)Wv[i]; }
}

// ---------------------------------------------------------------------------
// Kernel 2: projections.  One wave per (batch, 16-pixel tile).
// D[o,n] = sum_c W[o,c] * x[c,n] :  A = W tile (16x32), B = x tile (32x16).
// x is [B][C][N] f32 so B-fragment elements are per-lane strided scalar loads
// (coalesced across the 16 lanes sharing a c-row).
// ---------------------------------------------------------------------------
__global__ void __launch_bounds__(256) proj_kernel(
    const float* __restrict__ xa, const float* __restrict__ xb,
    const __bf16* __restrict__ wq, const __bf16* __restrict__ wk,
    const __bf16* __restrict__ wv,
    const float* __restrict__ bq, const float* __restrict__ bk,
    const float* __restrict__ bv,
    __bf16* __restrict__ Qb, __bf16* __restrict__ Kb, __bf16* __restrict__ Vt) {
  const int lane = threadIdx.x & 31;
  const int wid  = threadIdx.x >> 5;
  const int task = blockIdx.x * 8 + wid;      // B * N/16 = 1024 tasks
  const int bb   = task >> 8;
  const int nb   = (task & 255) * 16;
  const int col  = lane & 15;
  const int hh   = lane >> 4;
  const int n    = nb + col;

  v8f accq[2], acck[2], accv[16];
#pragma unroll
  for (int t = 0; t < 2; ++t) { accq[t] = (v8f){0,0,0,0,0,0,0,0}; acck[t] = (v8f){0,0,0,0,0,0,0,0}; }
#pragma unroll
  for (int t = 0; t < 16; ++t) accv[t] = (v8f){0,0,0,0,0,0,0,0};

#pragma unroll 1
  for (int cs = 0; cs < 8; ++cs) {            // K loop over C=256 in steps of 32
    const int c0 = cs * 32;
    v16bf bxa, bxb;
#pragma unroll
    for (int e = 0; e < 16; ++e) {            // B frag: element e <-> c = c0+16*hh+e
      const size_t idx = ((size_t)(bb * C_ + c0 + hh * 16 + e)) * N_ + n;
      bxa[e] = (__bf16)xa[idx];
      bxb[e] = (__bf16)xb[idx];
    }
#pragma unroll
    for (int ot = 0; ot < 2; ++ot) {
      accq[ot] = wmma_bf16(load_a16(wq + (size_t)(ot * 16 + col) * C_ + c0, hh), bxa, accq[ot]);
      acck[ot] = wmma_bf16(load_a16(wk + (size_t)(ot * 16 + col) * C_ + c0, hh), bxb, acck[ot]);
    }
#pragma unroll
    for (int ot = 0; ot < 16; ++ot) {
      accv[ot] = wmma_bf16(load_a16(wv + (size_t)(ot * 16 + col) * C_ + c0, hh), bxb, accv[ot]);
    }
  }

  // Epilogue: D layout -> lane col = n-local, VGPR r -> o_local = r + 8*hh.
#pragma unroll
  for (int ot = 0; ot < 2; ++ot)
#pragma unroll
    for (int r = 0; r < 8; ++r) {
      const int o = ot * 16 + r + 8 * hh;
      Qb[((size_t)(bb * N_ + n)) * CK_ + o] = (__bf16)(accq[ot][r] + bq[o]);
      Kb[((size_t)(bb * N_ + n)) * CK_ + o] = (__bf16)(acck[ot][r] + bk[o]);
    }
#pragma unroll
  for (int ot = 0; ot < 16; ++ot)
#pragma unroll
    for (int r = 0; r < 8; ++r) {
      const int o = ot * 16 + r + 8 * hh;
      Vt[((size_t)(bb * C_ + o)) * N_ + n] = (__bf16)(accv[ot][r] + bv[o]);
    }
}

// ---------------------------------------------------------------------------
// Kernel 3: flash attention + gamma-residual epilogue.
// Block = 512 threads = 16 waves: 8 query-row tiles (128 rows) x 2 channel
// halves (128 ch each).  Per wave: one 16x(CK=32) Q fragment held in regs,
// acc = 8 x v8f (16 rows x 128 channels), online softmax stats per D-row.
// ---------------------------------------------------------------------------
__global__ void __launch_bounds__(512) attn_kernel(
    const __bf16* __restrict__ Qb, const __bf16* __restrict__ Kb,
    const __bf16* __restrict__ Vt, const float* __restrict__ xa,
    const float* __restrict__ gamma, float* __restrict__ outp) {
  __shared__ __align__(16) unsigned char smem[16 * 1280];  // per-wave 1280B slice
  const int lane = threadIdx.x & 31;
  const int w    = threadIdx.x >> 5;   // 0..15
  const int rt   = w & 7;              // row tile within block
  const int ch   = w >> 3;             // channel half (0/1)
  const int col  = lane & 15;
  const int hh   = lane >> 4;
  const int bb   = blockIdx.x >> 5;    // 32 blocks per batch
  const int q0   = (blockIdx.x & 31) * 128 + rt * 16;

  __bf16* pl = (__bf16*)(smem + w * 1280);   // P stage: 16x32 bf16 (1KB)
  float*  ep = (float*)(smem + w * 1280);    // epilogue: 16x17 f32 (reuse)

  const float g0 = gamma[0];

  // Q A-fragment (rows q0..q0+15, K = CK = 32)
  const v16bf qa = load_a16(Qb + ((size_t)(bb * N_ + q0 + col)) * CK_, hh);

  v8f acc[8];
  float m_run[8], l_run[8];
#pragma unroll
  for (int t = 0; t < 8; ++t) acc[t] = (v8f){0,0,0,0,0,0,0,0};
#pragma unroll
  for (int r = 0; r < 8; ++r) { m_run[r] = -1e30f; l_run[r] = 0.0f; }

  const v8f zz = (v8f){0,0,0,0,0,0,0,0};

#pragma unroll 1
  for (int mc = 0; mc < N_; mc += 32) {      // stream keys in chunks of 32
    // scores: S[16q x 32k] via two WMMAs (B frag = k^T rows, contiguous)
    const __bf16* kp = Kb + ((size_t)(bb * N_ + mc + col)) * CK_ + hh * 16;
    const v16bf kb0 = load_b16(kp);
    const v16bf kb1 = load_b16(kp + 16 * CK_);
    // prefetch next chunk's K rows (L2 -> near cache) while we crunch
    if (mc + 32 < N_) __builtin_prefetch(kp + 32 * CK_, 0, 3);
    v8f s0 = wmma_bf16(qa, kb0, zz);
    v8f s1 = wmma_bf16(qa, kb1, zz);

    // online softmax per D-row (row m = r + 8*hh lives on 16 lanes of a VGPR)
#pragma unroll
    for (int r = 0; r < 8; ++r) {
      float tm = fmaxf(s0[r], s1[r]);
#pragma unroll
      for (int d = 8; d >= 1; d >>= 1) tm = fmaxf(tm, __shfl_xor(tm, d, 16));
      const float mn   = fmaxf(m_run[r], tm);
      const float corr = __expf(m_run[r] - mn);
      const float p0   = __expf(s0[r] - mn);
      const float p1   = __expf(s1[r] - mn);
      float rs = p0 + p1;
#pragma unroll
      for (int d = 8; d >= 1; d >>= 1) rs += __shfl_xor(rs, d, 16);
      l_run[r] = l_run[r] * corr + rs;
      m_run[r] = mn;
#pragma unroll
      for (int t = 0; t < 8; ++t) acc[t][r] *= corr;
      // stage P (D layout -> row-major [m][k] bf16 in LDS)
      pl[(r + 8 * hh) * 32 + col]      = (__bf16)p0;
      pl[(r + 8 * hh) * 32 + 16 + col] = (__bf16)p1;
    }
    __builtin_amdgcn_wave_barrier();
    // P A-fragment from LDS (intra-wave; DS ops are in-order per wave)
    const v16bf pa = load_a16(pl + col * 32, hh);
#pragma unroll
    for (int t = 0; t < 8; ++t) {
      const int cb = ch * 128 + t * 16;
      const __bf16* vp = Vt + ((size_t)(bb * C_ + cb + col)) * N_ + mc + hh * 16;
      const v16bf vb = load_b16(vp);
      if (mc + 32 < N_) __builtin_prefetch(vp + 32, 0, 3);  // next chunk's V rows
      acc[t] = wmma_bf16(pa, vb, acc[t]);
    }
    __builtin_amdgcn_wave_barrier();
  }

  // Epilogue: out[b,c,n] = x_a[b,c,n] + g * acc[n,c]/l[n], transposed through
  // LDS (pad 17) so global stores are 64B-coalesced per half-wave.
#pragma unroll
  for (int t = 0; t < 8; ++t) {
    const int cb = ch * 128 + t * 16;
    __builtin_amdgcn_wave_barrier();
#pragma unroll
    for (int r = 0; r < 8; ++r) {
      const int m = r + 8 * hh;                 // query row within tile
      ep[col * 17 + m] = acc[t][r] * (1.0f / l_run[r]);
    }
    __builtin_amdgcn_wave_barrier();
#pragma unroll
    for (int j = 0; j < 8; ++j) {
      const int cl = 2 * j + hh;                // channel local
      const float v = ep[cl * 17 + col];
      const size_t idx = ((size_t)(bb * C_ + cb + cl)) * N_ + (size_t)(q0 + col);
      outp[idx] = xa[idx] + g0 * v;
    }
  }
}

// ---------------------------------------------------------------------------
extern "C" void kernel_launch(void* const* d_in, const int* in_sizes, int n_in,
                              void* d_out, int out_size, void* d_ws, size_t ws_size,
                              hipStream_t stream) {
  (void)in_sizes; (void)n_in; (void)out_size; (void)ws_size;
  const float* xa    = (const float*)d_in[0];
  const float* xb    = (const float*)d_in[1];
  const float* Wq    = (const float*)d_in[2];
  const float* bq    = (const float*)d_in[3];
  const float* Wk    = (const float*)d_in[4];
  const float* bk    = (const float*)d_in[5];
  const float* Wv    = (const float*)d_in[6];
  const float* bv    = (const float*)d_in[7];
  const float* gamma = (const float*)d_in[8];
  float* outp = (float*)d_out;

  char* ws = (char*)d_ws;
  __bf16* Qb  = (__bf16*)(ws);
  __bf16* Kb  = (__bf16*)(ws + (1u << 20));
  __bf16* Vt  = (__bf16*)(ws + (2u << 20));
  __bf16* wqb = (__bf16*)(ws + (10u << 20));
  __bf16* wkb = (__bf16*)(ws + (10u << 20) + CK_ * C_ * 2);
  __bf16* wvb = (__bf16*)(ws + (10u << 20) + 2 * CK_ * C_ * 2);

  hipLaunchKernelGGL(convert_w_kernel, dim3(256), dim3(256), 0, stream,
                     Wq, Wk, Wv, wqb, wkb, wvb);
  hipLaunchKernelGGL(proj_kernel, dim3(128), dim3(256), 0, stream,
                     xa, xb, wqb, wkb, wvb, bq, bk, bv, Qb, Kb, Vt);
  hipLaunchKernelGGL(attn_kernel, dim3(B_ * 32), dim3(512), 0, stream,
                     Qb, Kb, Vt, xa, gamma, outp);
}
